// AttnBlock2d_76931454206447
// MI455X (gfx1250) — matrix operand order repaired and checked by reading, not compiled, not run on hardware
//
#include <hip/hip_runtime.h>
#include <hip/hip_bf16.h>

// ---------------------------------------------------------------------------
// AttnBlock2d for MI455X (gfx1250): GN -> QKV GEMM -> flash attention -> proj
// All heavy math on v_wmma_f32_16x16x32_bf16 (wave32, 16x16x32 tiles).
// Round 2: __launch_bounds__ added to kill accumulator spills-to-scratch;
// attention uses 4-wave (128-thread) blocks so each wave gets a full VGPR file.
// ---------------------------------------------------------------------------

typedef __attribute__((ext_vector_type(16))) __bf16 v16bf;
typedef __attribute__((ext_vector_type(8)))  __bf16 v8bf;
typedef __attribute__((ext_vector_type(8)))  float  v8f;

#define WMMA_BF16(a, b, c) \
  __builtin_amdgcn_wmma_f32_16x16x32_bf16(false, (a), false, (b), (short)0, (c), false, false)

static constexpr int BB = 8;
static constexpr int C  = 256;
static constexpr int HH = 64;
static constexpr int WW = 64;
static constexpr int NN = HH * WW;      // 4096 spatial positions
static constexpr int G  = 16;           // groups
static constexpr int CG = C / G;        // 16 channels/group

// ---- WMMA fragment loaders --------------------------------------------------
// A operand, 16(M) x 32(K) bf16, memory row-major [M][K], ld in elements.
// Lane l: row m = l&15; K rows kb=(l>=16?16:0): 16 contiguous K values.
__device__ __forceinline__ v16bf load_a_frag(const __bf16* base, int ld, int m0, int k0) {
  int lane = threadIdx.x & 31;
  int m  = m0 + (lane & 15);
  int kb = k0 + ((lane >> 4) << 4);
  union { v16bf v; v8bf h[2]; } u;
  u.h[0] = *(const v8bf*)(base + (size_t)m * ld + kb);
  u.h[1] = *(const v8bf*)(base + (size_t)m * ld + kb + 8);
  return u.v;
}

// B operand, 32(K) x 16(N) bf16, memory stored K-major-per-column: [N][K], ld in elems.
// Lane l: column n = l&15; K rows kb=(l>=16?16:0): 16 contiguous K values.
__device__ __forceinline__ v16bf load_b_frag(const __bf16* base, int ld, int k0, int n0) {
  int lane = threadIdx.x & 31;
  int n  = n0 + (lane & 15);
  int kb = k0 + ((lane >> 4) << 4);
  union { v16bf v; v8bf h[2]; } u;
  u.h[0] = *(const v8bf*)(base + (size_t)n * ld + kb);
  u.h[1] = *(const v8bf*)(base + (size_t)n * ld + kb + 8);
  return u.v;
}

// ---- 0) convert weights to bf16, fold C^-0.5 into Wq ------------------------
__global__ void __launch_bounds__(256) prep_kernel(
    const float* __restrict__ wq, const float* __restrict__ wk,
    const float* __restrict__ wv, const float* __restrict__ wp,
    __bf16* wqb, __bf16* wkb, __bf16* wvb, __bf16* wpb) {
  int i = blockIdx.x * blockDim.x + threadIdx.x;
  if (i < C * C) {
    wqb[i] = (__bf16)(wq[i] * 0.0625f);   // * C^-0.5 = 1/16
    wkb[i] = (__bf16)wk[i];
    wvb[i] = (__bf16)wv[i];
    wpb[i] = (__bf16)wp[i];
  }
}

// ---- 1) GroupNorm statistics: one workgroup per (b, g) ----------------------
__global__ void __launch_bounds__(256) gn_stats_kernel(
    const float* __restrict__ x, float* mean, float* rstd) {
  int bg = blockIdx.x;                 // 0..127
  int b = bg / G, g = bg % G;
  const float* xp = x + ((size_t)b * C + g * CG) * NN;
  float s = 0.f, sq = 0.f;
  for (int i = threadIdx.x; i < CG * NN; i += blockDim.x) {
    float v = xp[i];
    s += v; sq += v * v;
  }
  __shared__ float ls[256], lq[256];
  ls[threadIdx.x] = s; lq[threadIdx.x] = sq;
  __syncthreads();
  for (int off = 128; off > 0; off >>= 1) {
    if (threadIdx.x < off) {
      ls[threadIdx.x] += ls[threadIdx.x + off];
      lq[threadIdx.x] += lq[threadIdx.x + off];
    }
    __syncthreads();
  }
  if (threadIdx.x == 0) {
    float m   = ls[0] / (float)(CG * NN);
    float var = lq[0] / (float)(CG * NN) - m * m;
    mean[bg] = m;
    rstd[bg] = rsqrtf(var + 1e-6f);
  }
}

// ---- 2) apply GN, write h as bf16 [B][N][C] (transposed for B-operand) ------
__global__ void __launch_bounds__(256) gn_apply_kernel(
    const float* __restrict__ x, const float* __restrict__ gamma,
    const float* __restrict__ beta, const float* __restrict__ mean,
    const float* __restrict__ rstd, __bf16* __restrict__ hbf) {
  int bc = blockIdx.x;                 // 0..B*C-1
  int b = bc / C, c = bc % C;
  int bg = b * G + c / CG;
  float m = mean[bg], r = rstd[bg];
  float ga = gamma[c], be = beta[c];
  const float* xp = x + (size_t)bc * NN;
  __bf16* hp = hbf + (size_t)b * NN * C + c;
  for (int n = threadIdx.x; n < NN; n += blockDim.x)
    hp[(size_t)n * C] = (__bf16)((xp[n] - m) * r * ga + be);
}

// ---- 3) fused QKV GEMM: per wave one 16(co) x 64(n) tile, 3 outputs ---------
// q -> [B][N][C] (A-operand for attention), k -> [B][N][C] (B-operand),
// v -> [B][C][N] (B-operand of PV GEMM).
__global__ void __launch_bounds__(256, 1) qkv_kernel(
    const __bf16* __restrict__ wqb, const __bf16* __restrict__ wkb,
    const __bf16* __restrict__ wvb,
    const float* __restrict__ bq, const float* __restrict__ bk,
    const float* __restrict__ bv,
    const __bf16* __restrict__ hbf,
    __bf16* __restrict__ qb, __bf16* __restrict__ kbuf,
    __bf16* __restrict__ vbuf) {
  int wave = (blockIdx.x * blockDim.x + threadIdx.x) >> 5;
  int ngrp = wave % (NN / 64); wave /= (NN / 64);
  int cot  = wave % (C / 16);
  int b    = wave / (C / 16);
  int co0 = cot * 16, n0 = ngrp * 64;
  int lane = threadIdx.x & 31;

  v8f acc[3][4] = {};
  const __bf16* hB = hbf + (size_t)b * NN * C;
#pragma unroll 2
  for (int c0 = 0; c0 < C; c0 += 32) {
    v16bf aq = load_a_frag(wqb, C, co0, c0);
    v16bf ak = load_a_frag(wkb, C, co0, c0);
    v16bf av = load_a_frag(wvb, C, co0, c0);
#pragma unroll
    for (int t = 0; t < 4; ++t) {
      v16bf bf = load_b_frag(hB, C, c0, n0 + 16 * t);
      acc[0][t] = WMMA_BF16(aq, bf, acc[0][t]);
      acc[1][t] = WMMA_BF16(ak, bf, acc[1][t]);
      acc[2][t] = WMMA_BF16(av, bf, acc[2][t]);
    }
  }
  int nl = lane & 15, hl = lane >> 4;
#pragma unroll
  for (int r = 0; r < 8; ++r) {
    int co = co0 + r + 8 * hl;
    float biasq = bq[co] * 0.0625f;     // scale folded like Wq
    float biask = bk[co], biasv = bv[co];
#pragma unroll
    for (int t = 0; t < 4; ++t) {
      int n = n0 + 16 * t + nl;
      qb  [((size_t)b * NN + n) * C + co] = (__bf16)(acc[0][t][r] + biasq);
      kbuf[((size_t)b * NN + n) * C + co] = (__bf16)(acc[1][t][r] + biask);
      vbuf[((size_t)b * C + co) * NN + n] = (__bf16)(acc[2][t][r] + biasv);
    }
  }
}

// ---- 4) flash attention: wave = 16 queries x all 256 channels ---------------
// 4 waves / 128 threads per block with min-1-block bounds -> each wave can take
// the maximum VGPR allocation (O accum 128 + Q frags 64 + S 16 stay resident).
__global__ void __launch_bounds__(128, 1) attn_kernel(
    const __bf16* __restrict__ qb, const __bf16* __restrict__ kb,
    const __bf16* __restrict__ vb, __bf16* __restrict__ ob) {
  __shared__ __bf16 pstage[4][16 * 32];      // per-wave P tile staging (C/D -> A layout)
  int wv = threadIdx.x >> 5, lane = threadIdx.x & 31;
  int b    = blockIdx.x / (NN / 64);
  int qblk = blockIdx.x % (NN / 64);
  int i0 = qblk * 64 + wv * 16;

  const __bf16* qB = qb + (size_t)b * NN * C;
  const __bf16* kB = kb + (size_t)b * NN * C;
  const __bf16* vB = vb + (size_t)b * C * NN;

  v16bf qf[8];                                // Q fragments for all K=256, loaded once
#pragma unroll
  for (int kk = 0; kk < 8; ++kk) qf[kk] = load_a_frag(qB, C, i0, kk * 32);

  v8f o[16] = {};                             // O accum: 16 queries x 256 channels
  float mrow[8], ssum[8];
#pragma unroll
  for (int r = 0; r < 8; ++r) { mrow[r] = -1e30f; ssum[r] = 0.f; }

  __bf16* ps = pstage[wv];
  int nl = lane & 15, hl = lane >> 4;

  for (int j0 = 0; j0 < NN; j0 += 32) {
    // S = (Q * C^-0.5) . K^T  over K=256, two 16x16 j-subtiles
    v8f s0 = {}, s1 = {};
#pragma unroll
    for (int kk = 0; kk < 8; ++kk) {
      v16bf b0 = load_b_frag(kB, C, kk * 32, j0);
      v16bf b1 = load_b_frag(kB, C, kk * 32, j0 + 16);
      s0 = WMMA_BF16(qf[kk], b0, s0);
      s1 = WMMA_BF16(qf[kk], b1, s1);
    }
    // online softmax: row ops via shfl_xor inside 16-lane halves
    float corr[8];
#pragma unroll
    for (int r = 0; r < 8; ++r) {
      float mx = fmaxf(s0[r], s1[r]);
#pragma unroll
      for (int m = 1; m < 16; m <<= 1) mx = fmaxf(mx, __shfl_xor(mx, m, 32));
      float mnew = fmaxf(mrow[r], mx);
      corr[r] = __expf(mrow[r] - mnew);
      mrow[r] = mnew;
      float p0 = __expf(s0[r] - mnew);
      float p1 = __expf(s1[r] - mnew);
      s0[r] = p0; s1[r] = p1;
      float rs = p0 + p1;
#pragma unroll
      for (int m = 1; m < 16; m <<= 1) rs += __shfl_xor(rs, m, 32);
      ssum[r] = ssum[r] * corr[r] + rs;
    }
#pragma unroll
    for (int ct = 0; ct < 16; ++ct)
#pragma unroll
      for (int r = 0; r < 8; ++r) o[ct][r] *= corr[r];

    // stage P (C/D layout) into LDS as a row-major 16x32 bf16 tile
#pragma unroll
    for (int r = 0; r < 8; ++r) {
      int row = r + 8 * hl;
      ps[row * 32 + nl]      = (__bf16)s0[r];
      ps[row * 32 + 16 + nl] = (__bf16)s1[r];
    }
    __syncthreads();                          // uniform; orders DS write -> DS read

    // reload P in A-operand layout
    v16bf pf;
    {
      int m = lane & 15, kbse = hl * 16;
      union { v16bf v; v8bf h[2]; } u;
      u.h[0] = *(const v8bf*)(ps + m * 32 + kbse);
      u.h[1] = *(const v8bf*)(ps + m * 32 + kbse + 8);
      pf = u.v;
    }
    // O += P . V   (V stored [C][N]: lane's column c has contiguous j)
#pragma unroll
    for (int ct = 0; ct < 16; ++ct) {
      int cc = ct * 16 + nl;
      int jb = j0 + hl * 16;
      union { v16bf v; v8bf h[2]; } u;
      u.h[0] = *(const v8bf*)(vB + (size_t)cc * NN + jb);
      u.h[1] = *(const v8bf*)(vB + (size_t)cc * NN + jb + 8);
      o[ct] = WMMA_BF16(pf, u.v, o[ct]);
    }
    __syncthreads();
  }

  // normalize and store attn-out as bf16 [B][N][C] (B-operand for proj GEMM)
  __bf16* oB = ob + (size_t)b * NN * C;
  float inv[8];
#pragma unroll
  for (int r = 0; r < 8; ++r) inv[r] = 1.0f / ssum[r];
#pragma unroll
  for (int ct = 0; ct < 16; ++ct)
#pragma unroll
    for (int r = 0; r < 8; ++r) {
      int i = i0 + r + 8 * hl;
      oB[(size_t)i * C + ct * 16 + nl] = (__bf16)(o[ct][r] * inv[r]);
    }
}

// ---- 5) proj GEMM + bias + residual, f32 output -----------------------------
__global__ void __launch_bounds__(256, 1) proj_kernel(
    const __bf16* __restrict__ wpb, const float* __restrict__ bp,
    const __bf16* __restrict__ aob, const float* __restrict__ x,
    float* __restrict__ out) {
  int wave = (blockIdx.x * blockDim.x + threadIdx.x) >> 5;
  int ngrp = wave % (NN / 64); wave /= (NN / 64);
  int cot  = wave % (C / 16);
  int b    = wave / (C / 16);
  int co0 = cot * 16, n0 = ngrp * 64;

  v8f acc[4] = {};
  const __bf16* aB = aob + (size_t)b * NN * C;
#pragma unroll 2
  for (int c0 = 0; c0 < C; c0 += 32) {
    v16bf a = load_a_frag(wpb, C, co0, c0);
#pragma unroll
    for (int t = 0; t < 4; ++t) {
      v16bf bf = load_b_frag(aB, C, c0, n0 + 16 * t);
      acc[t] = WMMA_BF16(a, bf, acc[t]);
    }
  }
  int lane = threadIdx.x & 31, nl = lane & 15, hl = lane >> 4;
#pragma unroll
  for (int r = 0; r < 8; ++r) {
    int co = co0 + r + 8 * hl;
    float bias = bp[co];
#pragma unroll
    for (int t = 0; t < 4; ++t) {
      int n = n0 + 16 * t + nl;
      size_t idx = ((size_t)b * C + co) * NN + n;
      out[idx] = x[idx] + acc[t][r] + bias;
    }
  }
}

// ---------------------------------------------------------------------------
extern "C" void kernel_launch(void* const* d_in, const int* in_sizes, int n_in,
                              void* d_out, int out_size, void* d_ws, size_t ws_size,
                              hipStream_t stream) {
  const float* x     = (const float*)d_in[0];
  const float* gamma = (const float*)d_in[1];
  const float* beta  = (const float*)d_in[2];
  const float* wq    = (const float*)d_in[3];
  const float* bq    = (const float*)d_in[4];
  const float* wk    = (const float*)d_in[5];
  const float* bk    = (const float*)d_in[6];
  const float* wv    = (const float*)d_in[7];
  const float* bv    = (const float*)d_in[8];
  const float* wp    = (const float*)d_in[9];
  const float* bp    = (const float*)d_in[10];
  float* out = (float*)d_out;

  // workspace carve (~85 MB)
  char* w = (char*)d_ws;
  auto carve = [&](size_t bytes) { char* p = w; w += (bytes + 255) & ~(size_t)255; return p; };
  __bf16* wqb  = (__bf16*)carve((size_t)C * C * 2);
  __bf16* wkb  = (__bf16*)carve((size_t)C * C * 2);
  __bf16* wvb  = (__bf16*)carve((size_t)C * C * 2);
  __bf16* wpb  = (__bf16*)carve((size_t)C * C * 2);
  float*  mean = (float*)carve((size_t)BB * G * 4);
  float*  rstd = (float*)carve((size_t)BB * G * 4);
  __bf16* hbf  = (__bf16*)carve((size_t)BB * NN * C * 2);
  __bf16* qbf  = (__bf16*)carve((size_t)BB * NN * C * 2);
  __bf16* kbf  = (__bf16*)carve((size_t)BB * NN * C * 2);
  __bf16* vbf  = (__bf16*)carve((size_t)BB * C * NN * 2);
  __bf16* aobf = (__bf16*)carve((size_t)BB * NN * C * 2);

  prep_kernel<<<(C * C) / 256, 256, 0, stream>>>(wq, wk, wv, wp, wqb, wkb, wvb, wpb);
  gn_stats_kernel<<<BB * G, 256, 0, stream>>>(x, mean, rstd);
  gn_apply_kernel<<<BB * C, 256, 0, stream>>>(x, gamma, beta, mean, rstd, hbf);
  // waves = B * (C/16) * (N/64) = 8192 -> 1024 blocks of 8 waves
  qkv_kernel<<<1024, 256, 0, stream>>>(wqb, wkb, wvb, bq, bk, bv, hbf, qbf, kbf, vbf);
  // 4 waves (128 thr) per block, 16 queries/wave -> 64 queries/block
  attn_kernel<<<BB * (NN / 64), 128, 0, stream>>>(qbf, kbf, vbf, aobf);
  proj_kernel<<<1024, 256, 0, stream>>>(wpb, bp, aobf, x, out);
}
